// RNNModel_52158082843030
// MI455X (gfx1250) — matrix operand order repaired and checked by reading, not compiled, not run on hardware
//
#include <hip/hip_runtime.h>
#include <hip/hip_fp16.h>

#define T_STEPS 1500
#define BATCH   64
#define HID     512
#define FIN     36
#define FPAD    64
#define NCLS    64

typedef __attribute__((ext_vector_type(16))) _Float16 v16h;
typedef __attribute__((ext_vector_type(8)))  float    v8f;

union ABu { uint4 u4[2]; v16h v; };

// A-matrix (16x32 f16, row-major source with leading dim ld):
// lanes 0-15 : M=lane,   chunks K=[k0..k0+7]  and K=[k0+16..k0+23]
// lanes 16-31: M=lane-16, chunks K=[k0+8..k0+15] and K=[k0+24..k0+31]
__device__ __forceinline__ void loadA(const _Float16* __restrict__ src, int ld,
                                      int m0, int k0, int lane, ABu& a) {
  int row = m0 + (lane & 15);
  int c0  = k0 + ((lane >> 4) << 3);
  const _Float16* p = src + row * ld + c0;
  a.u4[0] = *reinterpret_cast<const uint4*>(p);
  a.u4[1] = *reinterpret_cast<const uint4*>(p + 16);
}

// B-matrix (32x16 f16) read from the block's LDS-resident weight slice.
// Slice layout: 64 rows (gate*16 + columnInSlice) x kld contiguous K.
// lane n (0-15): N=n, K=[k0..k0+7],[k0+16..k0+23]; lanes 16-31: N=lane-16, K=[k0+8..k0+15],[k0+24..k0+31]
__device__ __forceinline__ void loadB_lds(const _Float16* Wlds, int kld,
                                          int g, int k0, int lane, ABu& b) {
  int row = g * 16 + (lane & 15);
  int c0  = k0 + ((lane >> 4) << 3);
  const _Float16* p = Wlds + row * kld + c0;
  b.u4[0] = *reinterpret_cast<const uint4*>(p);
  b.u4[1] = *reinterpret_cast<const uint4*>(p + 16);
}

// Cooperative copy of this block's weight slice (4 gates x 16 cols x kld) into LDS.
__device__ __forceinline__ void stage_weights(const _Float16* __restrict__ W, int kld,
                                              int n0, _Float16* dst) {
  const int cpr    = kld >> 3;        // 16-byte chunks per row
  const int chunks = 64 * cpr;
  for (int c = threadIdx.x; c < chunks; c += blockDim.x) {
    int r    = c / cpr;
    int k8   = (c - r * cpr) << 3;
    int grow = (r >> 4) * HID + n0 + (r & 15);   // gate*HID + column
    *reinterpret_cast<uint4*>(dst + r * kld + k8) =
        *reinterpret_cast<const uint4*>(W + grow * kld + k8);
  }
}

__device__ __forceinline__ float sigf(float x) { return 1.f / (1.f + __expf(-x)); }

// Persistent two-layer pipelined LSTM.
// Blocks 0..31  -> layer 0, n-slice = blockIdx.x*16   (columns of H)
// Blocks 32..63 -> layer 1, n-slice = (blockIdx.x-32)*16
// 4 waves per block, wave w owns M-tile rows [16w..16w+15] of the batch.
// Weight slices live in LDS for the whole sequence; cell state c lives in VGPRs.
__global__ __launch_bounds__(128, 1)
void lstm_persistent(const float* __restrict__ x,
                     const _Float16* __restrict__ Wih0,   // 2048 x 64 (padded)
                     const _Float16* __restrict__ Whh0,   // 2048 x 512
                     const _Float16* __restrict__ Wih1,   // 2048 x 512
                     const _Float16* __restrict__ Whh1,   // 2048 x 512
                     const float* __restrict__ bias0,     // 2048 (b_ih+b_hh)
                     const float* __restrict__ bias1,
                     _Float16* __restrict__ h0buf,        // 2 x 64 x 512 ping-pong
                     _Float16* __restrict__ h1buf,        // 2 x 64 x 512 ping-pong
                     _Float16* __restrict__ xpad,         // 2 x 64 x 64 ping-pong
                     float*    __restrict__ h1last,       // 64 x 512 fp32
                     unsigned* __restrict__ sync) {
  extern __shared__ _Float16 lds[];

  const int  lane = threadIdx.x & 31;
  const int  wave = threadIdx.x >> 5;
  const int  m0   = wave * 16;
  const bool isL1 = blockIdx.x >= 32;
  const int  n0   = (isL1 ? (int)blockIdx.x - 32 : (int)blockIdx.x) * 16;
  const int  colN = n0 + (lane & 15);

  const float* bias = isL1 ? bias1 : bias0;
  const float b_i = bias[0 * HID + colN];
  const float b_f = bias[1 * HID + colN];
  const float b_g = bias[2 * HID + colN];
  const float b_o = bias[3 * HID + colN];

  const _Float16* Wi  = isL1 ? Wih1 : Wih0;
  const _Float16* Wh  = isL1 ? Whh1 : Whh0;
  const int       kin = isL1 ? HID : FPAD;

  // One-time LDS staging of this block's weight slice.
  _Float16* ldsWi = lds;                 // 64 x kin
  _Float16* ldsWh = lds + 64 * kin;      // 64 x 512
  stage_weights(Wi, kin, n0, ldsWi);
  stage_weights(Wh, HID, n0, ldsWh);
  __syncthreads();

  float c8[8];
#pragma unroll
  for (int r = 0; r < 8; ++r) c8[r] = 0.f;

  unsigned* cnt = sync;
  unsigned* gen = sync + 32;  // separate cacheline

  const v8f zero8 = {0.f, 0.f, 0.f, 0.f, 0.f, 0.f, 0.f, 0.f};

  for (int tick = 0; tick <= T_STEPS; ++tick) {
    const bool active = isL1 ? (tick >= 1) : (tick < T_STEPS);
    if (active) {
      const int cur = tick & 1, prv = cur ^ 1;
      // layer0 input: staged x slice for t=tick; layer1 input: layer0 h of t=tick-1
      const _Float16* inA  = isL1 ? (h0buf + prv * (BATCH * HID))
                                  : (xpad  + cur * (BATCH * FPAD));
      const int       inld = isL1 ? HID : FPAD;
      const _Float16* hprev = (isL1 ? h1buf : h0buf) + prv * (BATCH * HID);
      _Float16*       hcur  = (isL1 ? h1buf : h0buf) + cur * (BATCH * HID);

      v8f acc[4] = {zero8, zero8, zero8, zero8};
      ABu a, b;

      // input-to-gate GEMM contribution (weights from LDS)
      for (int k0 = 0; k0 < kin; k0 += 32) {
        loadA(inA, inld, m0, k0, lane, a);
#pragma unroll
        for (int g = 0; g < 4; ++g) {
          loadB_lds(ldsWi, kin, g, k0, lane, b);
          acc[g] = __builtin_amdgcn_wmma_f32_16x16x32_f16(
              false, a.v, false, b.v, (short)0, acc[g], false, false);
        }
      }
      // hidden-to-gate GEMM contribution (weights from LDS)
      for (int k0 = 0; k0 < HID; k0 += 32) {
        loadA(hprev, HID, m0, k0, lane, a);
#pragma unroll
        for (int g = 0; g < 4; ++g) {
          loadB_lds(ldsWh, HID, g, k0, lane, b);
          acc[g] = __builtin_amdgcn_wmma_f32_16x16x32_f16(
              false, a.v, false, b.v, (short)0, acc[g], false, false);
        }
      }

      // LSTM cell update, entirely in registers (C/D layout:
      // VGPR r, lanes 0-15 -> M=r, N=lane; lanes 16-31 -> M=8+r, N=lane-16)
#pragma unroll
      for (int r = 0; r < 8; ++r) {
        float gi = sigf(acc[0][r] + b_i);
        float gf = sigf(acc[1][r] + b_f);
        float gg = tanhf(acc[2][r] + b_g);
        float go = sigf(acc[3][r] + b_o);
        float cn = gf * c8[r] + gi * gg;
        c8[r] = cn;
        float hv = go * tanhf(cn);
        int row = m0 + r + ((lane >> 4) << 3);
        hcur[row * HID + colN] = (_Float16)hv;
        if (isL1 && tick == T_STEPS) h1last[row * HID + colN] = hv;
      }
    }

    // stage next x slice (block 0 only) into the other ping-pong half
    if (blockIdx.x == 0 && (tick + 1) < T_STEPS) {
      _Float16* dst = xpad + ((tick + 1) & 1) * (BATCH * FPAD);
      for (int idx = threadIdx.x; idx < BATCH * FPAD; idx += blockDim.x) {
        int bb = idx >> 6, ff = idx & 63;
        float v = (ff < FIN) ? x[(bb * T_STEPS + (tick + 1)) * FIN + ff] : 0.f;
        dst[idx] = (_Float16)v;
      }
    }

    // device-wide barrier (sense via monotonically increasing generation)
    __syncthreads();
    if (threadIdx.x == 0) {
      unsigned g0 = __hip_atomic_load(gen, __ATOMIC_ACQUIRE, __HIP_MEMORY_SCOPE_AGENT);
      unsigned a0 = __hip_atomic_fetch_add(cnt, 1u, __ATOMIC_ACQ_REL, __HIP_MEMORY_SCOPE_AGENT);
      if (a0 == gridDim.x - 1) {
        __hip_atomic_store(cnt, 0u, __ATOMIC_RELAXED, __HIP_MEMORY_SCOPE_AGENT);
        __hip_atomic_fetch_add(gen, 1u, __ATOMIC_ACQ_REL, __HIP_MEMORY_SCOPE_AGENT);
      } else {
        while (__hip_atomic_load(gen, __ATOMIC_ACQUIRE, __HIP_MEMORY_SCOPE_AGENT) == g0) {
          __builtin_amdgcn_s_sleep(8);
        }
      }
    }
    __syncthreads();
  }
}

// ---------- prep / epilogue kernels ----------

__global__ void k_f2h(const float* __restrict__ s, _Float16* __restrict__ d, int n) {
  int i = blockIdx.x * blockDim.x + threadIdx.x;
  if (i < n) d[i] = (_Float16)s[i];
}

__global__ void k_wih0_pad(const float* __restrict__ s, _Float16* __restrict__ d) {
  int i = blockIdx.x * blockDim.x + threadIdx.x;           // 2048 x 64
  if (i >= 4 * HID * FPAD) return;
  int row = i >> 6, ff = i & 63;
  d[i] = (_Float16)((ff < FIN) ? s[row * FIN + ff] : 0.f);
}

__global__ void k_bias(const float* __restrict__ bi, const float* __restrict__ bh,
                       float* __restrict__ d, int n) {
  int i = blockIdx.x * blockDim.x + threadIdx.x;
  if (i < n) d[i] = bi[i] + bh[i];
}

__global__ void k_init(const float* __restrict__ x, _Float16* __restrict__ h0,
                       _Float16* __restrict__ h1, _Float16* __restrict__ xp,
                       unsigned* __restrict__ sync) {
  int i = blockIdx.x * blockDim.x + threadIdx.x;
  if (i < 2 * BATCH * HID) { h0[i] = (_Float16)0.f; h1[i] = (_Float16)0.f; }
  if (i < BATCH * FPAD) {
    int bb = i >> 6, ff = i & 63;
    xp[i] = (_Float16)((ff < FIN) ? x[(bb * T_STEPS) * FIN + ff] : 0.f);
  }
  if (i == 0) sync[0] = 0u;
}

__global__ void k_out(const float* __restrict__ h1last, const float* __restrict__ Wout,
                      const float* __restrict__ bout, float* __restrict__ out) {
  int i = blockIdx.x * blockDim.x + threadIdx.x;
  if (i >= BATCH * NCLS) return;
  int bb = i >> 6, cc = i & 63;
  float s = bout[cc];
  const float* hp = h1last + bb * HID;
  const float* wp = Wout + cc * HID;
  for (int k = 0; k < HID; ++k) s += hp[k] * wp[k];
  out[i] = s;
}

extern "C" void kernel_launch(void* const* d_in, const int* in_sizes, int n_in,
                              void* d_out, int out_size, void* d_ws, size_t ws_size,
                              hipStream_t stream) {
  (void)in_sizes; (void)n_in; (void)out_size; (void)ws_size;
  const float* x     = (const float*)d_in[0];
  const float* Wih0f = (const float*)d_in[1];
  const float* Whh0f = (const float*)d_in[2];
  const float* bih0  = (const float*)d_in[3];
  const float* bhh0  = (const float*)d_in[4];
  const float* Wih1f = (const float*)d_in[5];
  const float* Whh1f = (const float*)d_in[6];
  const float* bih1  = (const float*)d_in[7];
  const float* bhh1  = (const float*)d_in[8];
  const float* Woutf = (const float*)d_in[9];
  const float* boutf = (const float*)d_in[10];
  float* out = (float*)d_out;

  char* w = (char*)d_ws;
  auto carve = [&](size_t bytes) -> void* {
    void* p = (void*)w;
    w += (bytes + 255) & ~(size_t)255;
    return p;
  };
  _Float16* Wih0h  = (_Float16*)carve((size_t)4 * HID * FPAD * 2);
  _Float16* Whh0h  = (_Float16*)carve((size_t)4 * HID * HID * 2);
  _Float16* Wih1h  = (_Float16*)carve((size_t)4 * HID * HID * 2);
  _Float16* Whh1h  = (_Float16*)carve((size_t)4 * HID * HID * 2);
  float*    bias0  = (float*)carve((size_t)4 * HID * 4);
  float*    bias1  = (float*)carve((size_t)4 * HID * 4);
  _Float16* h0buf  = (_Float16*)carve((size_t)2 * BATCH * HID * 2);
  _Float16* h1buf  = (_Float16*)carve((size_t)2 * BATCH * HID * 2);
  _Float16* xpad   = (_Float16*)carve((size_t)2 * BATCH * FPAD * 2);
  float*    h1last = (float*)carve((size_t)BATCH * HID * 4);
  unsigned* sync   = (unsigned*)carve(256);

  const int nW = 4 * HID * HID;  // 1048576
  k_f2h<<<(nW + 255) / 256, 256, 0, stream>>>(Whh0f, Whh0h, nW);
  k_f2h<<<(nW + 255) / 256, 256, 0, stream>>>(Wih1f, Wih1h, nW);
  k_f2h<<<(nW + 255) / 256, 256, 0, stream>>>(Whh1f, Whh1h, nW);
  k_wih0_pad<<<(4 * HID * FPAD + 255) / 256, 256, 0, stream>>>(Wih0f, Wih0h);
  k_bias<<<(4 * HID + 255) / 256, 256, 0, stream>>>(bih0, bhh0, bias0, 4 * HID);
  k_bias<<<(4 * HID + 255) / 256, 256, 0, stream>>>(bih1, bhh1, bias1, 4 * HID);
  k_init<<<(2 * BATCH * HID + 255) / 256, 256, 0, stream>>>(x, h0buf, h1buf, xpad, sync);

  // 128 KB dynamic LDS: layer-1 blocks stage 64x512 + 64x512 fp16 weight slices.
  const size_t ldsBytes = (size_t)(64 * HID + 64 * HID) * sizeof(_Float16);
  lstm_persistent<<<64, 128, ldsBytes, stream>>>(x, Wih0h, Whh0h, Wih1h, Whh1h,
                                                 bias0, bias1, h0buf, h1buf, xpad,
                                                 h1last, sync);

  k_out<<<(BATCH * NCLS + 255) / 256, 256, 0, stream>>>(h1last, Woutf, boutf, out);
}
